// GraphFPSumModule_11690900979989
// MI455X (gfx1250) — matrix operand order repaired and checked by reading, not compiled, not run on hardware
//
#include <hip/hip_runtime.h>
#include <hip/hip_bf16.h>
#include <math.h>

// ---------------- problem constants ----------------
#define BSZ   2
#define NPTS  4096
#define MPTS  1024
#define NT_   (BSZ*NPTS)     // 8192 fine nodes
#define MT_   (BSZ*MPTS)     // 2048 coarse nodes
#define CCH   32             // channels per degree
#define CMIDC 8              // key/value channels
#define KEDGE 12             // incoming edges per node (dst-major)
#define ETOT  (NT_*KEDGE)    // 98304 edges
#define TE    16             // edges per tile == one wave's WMMA M dim

typedef __attribute__((ext_vector_type(16))) __bf16 v16bf;
typedef __attribute__((ext_vector_type(8)))  float  v8f;

// Radial-MLP params: raw f32 (biases/LN) + pre-packed bf16 weight fragments in ws.
struct RadialP {
  const float *w1,*b1,*g1,*be1,*b2,*g2,*be2,*b3;
  const __bf16 *w2p, *w3p;     // fragment-layout packed: [tile][lane(32)][16 bf16]
};
struct EdgeArgs { RadialP rp[2][4]; };   // [0]='v' params, [1]='k' params; combos 00,01,10,11

struct PackDesc { const float* src; int ncols; int ntiles; unsigned long dstoff; };
struct PackArgs { PackDesc d[16]; };

// ---------------- WMMA helpers (gfx1250, wave32) ----------------
__device__ inline v8f wmma_bf16(v16bf a, v16bf b, v8f c) {
  // D = A(16x32 bf16) * B(32x16 bf16) + C(16x16 f32)
  return __builtin_amdgcn_wmma_f32_16x16x32_bf16(false, a, false, b, (short)0, c,
                                                 false, false);
}

// A fragment: lane l holds row M=l; half hi: element j -> K=hi*16+j
__device__ inline v16bf load_afrag(const float (*sA)[CCH], int l, int hi) {
  v16bf a;
#pragma unroll
  for (int j = 0; j < 8; ++j) a[j]     = (__bf16)sA[l][hi*16 + j];
#pragma unroll
  for (int j = 0; j < 8; ++j) a[8 + j] = (__bf16)sA[l][hi*16 + 8 + j];
  return a;
}

// LayerNorm(32)+ReLU for one 16-edge tile: lanes (l, l+16) each own 16 channels of row l.
__device__ inline void ln_relu_row(const float* __restrict__ rin, float* __restrict__ rout,
                                   const float* __restrict__ g, const float* __restrict__ be,
                                   int hi) {
  float v[16]; float s = 0.f, s2 = 0.f;
#pragma unroll
  for (int mm = 0; mm < 16; ++mm) { float t = rin[hi*16 + mm]; v[mm] = t; s += t; s2 += t*t; }
  s  += __shfl_xor(s, 16, 32);
  s2 += __shfl_xor(s2, 16, 32);
  float mu  = s  * (1.f/32.f);
  float var = s2 * (1.f/32.f) - mu*mu;
  float inv = rsqrtf(var + 1e-5f);
#pragma unroll
  for (int mm = 0; mm < 16; ++mm) {
    int m = hi*16 + mm;
    rout[m] = fmaxf((v[mm] - mu)*inv*g[m] + be[m], 0.f);
  }
}

__device__ inline float wave_sum(float v) {
#pragma unroll
  for (int off = 1; off < 32; off <<= 1) v += __shfl_xor(v, off, 32);
  return v;
}

// ========= Kernel 0: pack w2/w3 into bf16 WMMA B-fragment layout (one shot, 208 tiles) =========
// Fragment element j of lane (l,hi) for column tile t = W[K = hi*16+j][t*16 + l].
__global__ __launch_bounds__(32) void pack_weights_kernel(PackArgs A, __bf16* __restrict__ dst)
{
  int b = blockIdx.x;
  const int lane = threadIdx.x, l = lane & 15, hi = lane >> 4;
  int mi = 0, t = b;
  while (mi < 16 && t >= A.d[mi].ntiles) { t -= A.d[mi].ntiles; ++mi; }
  if (mi >= 16) return;
  const PackDesc D = A.d[mi];
  const int col = t*16 + l;
  __bf16* o = dst + D.dstoff + ((size_t)t*32 + lane)*16;
#pragma unroll
  for (int j = 0; j < 16; ++j) o[j] = (__bf16)D.src[(hi*16 + j)*D.ncols + col];
}

// ================= Kernel 1: 3-NN inverse-distance interp + skip add =================
__global__ __launch_bounds__(128) void interp_kernel(
    const float* __restrict__ h0, const float* __restrict__ h1,
    const float* __restrict__ uph0, const float* __restrict__ uph1,
    const float* __restrict__ x, const float* __restrict__ xprev,
    float* __restrict__ hs0, float* __restrict__ hs1)
{
  __shared__ float sxp[MPTS*3];
  const int tid = threadIdx.x;
  const long n  = (long)blockIdx.x*128 + tid;
  const int  b  = (int)(n / NPTS);           // 128 | 4096, block stays in one batch
  for (int i = tid; i < MPTS*3; i += 128) sxp[i] = xprev[(long)b*MPTS*3 + i];
  __syncthreads();
  const float px = x[n*3], py = x[n*3+1], pz = x[n*3+2];
  float d0 = 3.4e38f, d1 = 3.4e38f, d2 = 3.4e38f; int i0 = 0, i1 = 0, i2 = 0;
  for (int j = 0; j < MPTS; ++j) {
    float dx = px - sxp[j*3], dy = py - sxp[j*3+1], dz = pz - sxp[j*3+2];
    float d = dx*dx + dy*dy + dz*dz;
    if (d < d0)      { d2=d1; i2=i1; d1=d0; i1=i0; d0=d; i0=j; }
    else if (d < d1) { d2=d1; i2=i1; d1=d;  i1=j; }
    else if (d < d2) { d2=d;  i2=j; }
  }
  d0 = fmaxf(d0, 1e-10f); d1 = fmaxf(d1, 1e-10f); d2 = fmaxf(d2, 1e-10f);
  float w0 = 1.f/(d0 + 1e-8f), w1 = 1.f/(d1 + 1e-8f), w2 = 1.f/(d2 + 1e-8f);
  float wsum = w0 + w1 + w2; w0 /= wsum; w1 /= wsum; w2 /= wsum;
  const long r0 = (long)b*MPTS + i0, r1 = (long)b*MPTS + i1, r2 = (long)b*MPTS + i2;
  for (int c = 0; c < CCH; ++c)
    hs0[n*CCH + c] = h0[n*CCH + c] + w0*uph0[r0*CCH + c] + w1*uph0[r1*CCH + c]
                                   + w2*uph0[r2*CCH + c];
  for (int c = 0; c < CCH*3; ++c)
    hs1[n*96 + c] = h1[n*96 + c] + w0*uph1[r0*96 + c] + w1*uph1[r1*96 + c]
                                 + w2*uph1[r2*96 + c];
}

// ======= Kernel 2: per-edge radial MLPs (bf16 WMMA) + basis contraction -> msg_k/msg_v =======
__global__ __launch_bounds__(32) void edge_kernel(
    const float* __restrict__ hs0g, const float* __restrict__ hs1g,
    const float* __restrict__ rg,
    const float* __restrict__ b00g, const float* __restrict__ b01g,
    const float* __restrict__ b10g, const float* __restrict__ b11g,
    const int*   __restrict__ srcg,
    float* __restrict__ msgv, float* __restrict__ msgk,
    EdgeArgs P)
{
  __shared__ float ss0[TE][CCH];          // gathered deg-0 source features
  __shared__ float ss1[TE][CCH*3];        // gathered deg-1 source features
  __shared__ float sb00[TE], sb01[TE][3], sb10[TE][3], sb11[TE][27];
  __shared__ float tbuf[TE][288];         // t[e, (i*q+qq)*nf+f]
  __shared__ float sA[TE][CCH];           // WMMA A staging (activations)
  __shared__ float hbuf[TE][CCH];         // pre-LN staging
  __shared__ float rtile[TE][96];         // one o-block of R (+bias)
  __shared__ float mv[TE][32], mk[TE][32];
  __shared__ float sr[TE];
  __shared__ int   ssrc[TE];

  const int lane = threadIdx.x;
  const int l = lane & 15, hi = lane >> 4;
  const int e0 = blockIdx.x * TE;

  if (lane < TE) { ssrc[lane] = srcg[e0 + lane]; sr[lane] = rg[e0 + lane];
                   sb00[lane] = b00g[e0 + lane]; }
  for (int idx = lane; idx < TE*3; idx += 32) { int e = idx/3, j = idx%3;
    sb01[e][j] = b01g[(e0+e)*3 + j]; sb10[e][j] = b10g[(e0+e)*3 + j]; }
  for (int idx = lane; idx < TE*27; idx += 32) { int e = idx/27, j = idx%27;
    sb11[e][j] = b11g[(e0+e)*27 + j]; }
  for (int idx = lane; idx < TE*32; idx += 32) { int e = idx>>5, c = idx&31;
    mv[e][c] = 0.f; mk[e][c] = 0.f; }
  __syncthreads();
  for (int idx = lane; idx < TE*32; idx += 32) { int e = idx>>5, c = idx&31;
    ss0[e][c] = hs0g[(long)ssrc[e]*CCH + c]; }
  for (int idx = lane; idx < TE*96; idx += 32) { int e = idx/96, c = idx%96;
    ss1[e][c] = hs1g[(long)ssrc[e]*96 + c]; }
  __syncthreads();

  const int cdi[4] = {0,0,1,1};
  const int cdo[4] = {0,1,0,1};
  const int cq [4] = {1,3,1,3};
  const int cp [4] = {1,1,3,3};
  const int cnf[4] = {1,1,1,3};

  for (int cb = 0; cb < 4; ++cb) {
    const int di = cdi[cb], dout = cdo[cb], q = cq[cb], p = cp[cb], nf = cnf[cb];
    const int qn = q*nf;
    // ---- t[e,i,qq,f] = sum_p basis[e,qq,p,f] * s[e,i,p] ----
    for (int idx = lane; idx < TE*32*qn; idx += 32) {
      int e = idx / (32*qn), rem = idx % (32*qn);
      int i = rem / qn, qf = rem % qn, qq = qf / nf, f = qf % nf;
      float acc = 0.f;
      for (int pp = 0; pp < p; ++pp) {
        float bb;
        if (cb == 0)      bb = sb00[e];
        else if (cb == 1) bb = sb01[e][qq];
        else if (cb == 2) bb = sb10[e][pp];
        else              bb = sb11[e][qq*9 + pp*3 + f];
        float sv = (di == 0) ? ss0[e][i] : ss1[e][i*3 + pp];
        acc += bb * sv;
      }
      tbuf[e][rem] = acc;
    }
    __syncthreads();

    for (int vk = 0; vk < 2; ++vk) {               // 0 -> 'v' path, 1 -> 'k' path
      const RadialP rp = P.rp[vk][cb];
      // ---- layer1: r*w1 + b1 -> LN -> ReLU ----
      for (int idx = lane; idx < TE*32; idx += 32) { int e = idx>>5, m = idx&31;
        hbuf[e][m] = sr[e]*rp.w1[m] + rp.b1[m]; }
      __syncthreads();
      ln_relu_row(hbuf[l], sA[l], rp.g1, rp.be1, hi);
      __syncthreads();
      // ---- layer2: [16x32]@[32x32] in 2 WMMA, +b2 -> LN -> ReLU ----
      v16bf a = load_afrag(sA, l, hi);
#pragma unroll
      for (int nt = 0; nt < 2; ++nt) {
        v16bf b = *reinterpret_cast<const v16bf*>(rp.w2p + ((size_t)nt*32 + lane)*16);
        v8f c;
#pragma unroll
        for (int vv = 0; vv < 8; ++vv) c[vv] = 0.f;
        c = wmma_bf16(a, b, c);
        float bias = rp.b2[nt*16 + l];
#pragma unroll
        for (int vv = 0; vv < 8; ++vv) hbuf[vv + hi*8][nt*16 + l] = c[vv] + bias;
      }
      __syncthreads();
      ln_relu_row(hbuf[l], sA[l], rp.g2, rp.be2, hi);
      __syncthreads();
      // ---- layer3: [16x32]@[32x(256*nf)] fused with basis contraction ----
      a = load_afrag(sA, l, hi);
      for (int o = 0; o < CMIDC; ++o) {
        for (int nt = 0; nt < 2*nf; ++nt) {
          const int tIdx = o*(2*nf) + nt;
          __builtin_prefetch(rp.w3p + ((size_t)(tIdx + 1)*32 + lane)*16, 0, 1);
          v16bf b = *reinterpret_cast<const v16bf*>(rp.w3p + ((size_t)tIdx*32 + lane)*16);
          v8f c;
#pragma unroll
          for (int vv = 0; vv < 8; ++vv) c[vv] = 0.f;
          c = wmma_bf16(a, b, c);
          float bias = rp.b3[o*32*nf + nt*16 + l];
#pragma unroll
          for (int vv = 0; vv < 8; ++vv) rtile[vv + hi*8][nt*16 + l] = c[vv] + bias;
        }
        __syncthreads();
        // msg[e,o,qq] += sum_{i,f} R[e,(o,i,f)] * t[e,i,qq,f]
        for (int idx = lane; idx < TE*q; idx += 32) {
          int e = idx / q, qq = idx % q;
          float acc = 0.f;
          for (int cc = 0; cc < 32*nf; ++cc) {
            int i = cc / nf, f = cc % nf;
            acc += rtile[e][cc] * tbuf[e][(i*q + qq)*nf + f];
          }
          float* mrow = (vk == 0) ? mv[e] : mk[e];
          int off = (dout == 0) ? o : (8 + o*3 + qq);
          mrow[off] += acc;
        }
        __syncthreads();
      }
    }
    __syncthreads();
  }
  for (int idx = lane; idx < TE*32; idx += 32) { int e = idx>>5, c = idx&31;
    msgv[(long)(e0 + e)*32 + c] = mv[e][c];
    msgk[(long)(e0 + e)*32 + c] = mk[e][c];
  }
}

// ===== Kernel 3: per-node q-proj, edge softmax, weighted sum, cat-proj, GNormSE3 =====
__global__ __launch_bounds__(128) void node_kernel(
    const float* __restrict__ hs0g, const float* __restrict__ hs1g,
    const float* __restrict__ msgv, const float* __restrict__ msgk,
    const float* __restrict__ qw0, const float* __restrict__ qw1,
    const float* __restrict__ pw0, const float* __restrict__ pw1,
    const float* __restrict__ ng0, const float* __restrict__ nb0,
    const float* __restrict__ ng1, const float* __restrict__ nb1,
    float* __restrict__ out)
{
  __shared__ float shs0[4][32], shs1[4][96], sq[4][32], satt[4][KEDGE], szz[4][32];
  const int w = threadIdx.x >> 5, lane = threadIdx.x & 31;
  const int n = blockIdx.x*4 + w;
  shs0[w][lane] = hs0g[(long)n*32 + lane];
  for (int i = lane; i < 96; i += 32) shs1[w][i] = hs1g[(long)n*96 + i];
  __syncthreads();
  // q projection, packed [q0(8) | q1(o*3+m, 24)] to match msg_k layout
  float qv = 0.f;
  if (lane < 8) { for (int c = 0; c < 32; ++c) qv += qw0[lane*32 + c]*shs0[w][c]; }
  else { int o = (lane - 8)/3, m = (lane - 8)%3;
         for (int c = 0; c < 32; ++c) qv += qw1[o*32 + c]*shs1[w][c*3 + m]; }
  sq[w][lane] = qv;
  __syncthreads();
  if (lane < KEDGE) {
    long e = (long)n*KEDGE + lane;
    float acc = 0.f;
    for (int c = 0; c < 32; ++c) acc += msgk[e*32 + c]*sq[w][c];
    satt[w][lane] = acc * 0.17677669529663687f;   // 1/sqrt(32)
  }
  __syncthreads();
  float mx = -3.4e38f;
  for (int k = 0; k < KEDGE; ++k) mx = fmaxf(mx, satt[w][k]);
  float den = 0.f;
  for (int k = 0; k < KEDGE; ++k) den += expf(satt[w][k] - mx);
  float z = 0.f;
  for (int k = 0; k < KEDGE; ++k) {
    float a = expf(satt[w][k] - mx) / den;
    z += a * msgv[((long)n*KEDGE + k)*32 + lane];
  }
  szz[w][lane] = z;
  __syncthreads();
  // cat-projection: lane = output channel o, cat = [z(8) | hs(32)] = 40 channels
  float h20 = 0.f, h21[3] = {0.f, 0.f, 0.f};
  for (int ch = 0; ch < 40; ++ch) {
    float c0v = (ch < 8) ? szz[w][ch] : shs0[w][ch - 8];
    h20 += pw0[lane*40 + ch]*c0v;
#pragma unroll
    for (int m = 0; m < 3; ++m) {
      float c1v = (ch < 8) ? szz[w][8 + ch*3 + m] : shs1[w][(ch - 8)*3 + m];
      h21[m] += pw1[lane*40 + ch]*c1v;
    }
  }
  // GNormSE3 deg-0
  float n0 = fmaxf(fabsf(h20), 1e-12f);
  float mu  = wave_sum(n0)      *(1.f/32.f);
  float var = wave_sum(n0*n0)   *(1.f/32.f) - mu*mu;
  float t0 = fmaxf((n0 - mu)*rsqrtf(var + 1e-5f)*ng0[lane] + nb0[lane], 0.f);
  out[(long)n*32 + lane] = t0*(h20/n0);
  // GNormSE3 deg-1
  float n1 = sqrtf(h21[0]*h21[0] + h21[1]*h21[1] + h21[2]*h21[2]);
  n1 = fmaxf(n1, 1e-12f);
  mu  = wave_sum(n1)    *(1.f/32.f);
  var = wave_sum(n1*n1) *(1.f/32.f) - mu*mu;
  float t1 = fmaxf((n1 - mu)*rsqrtf(var + 1e-5f)*ng1[lane] + nb1[lane], 0.f);
#pragma unroll
  for (int m = 0; m < 3; ++m)
    out[(long)NT_*32 + (long)n*96 + lane*3 + m] = t1*(h21[m]/n1);
}

// ============================== launch ==============================
extern "C" void kernel_launch(void* const* d_in, const int* in_sizes, int n_in,
                              void* d_out, int out_size, void* d_ws, size_t ws_size,
                              hipStream_t stream)
{
  (void)in_sizes; (void)n_in; (void)out_size; (void)ws_size;
  // setup_inputs() insertion order, params flattened depth-first in insertion order:
  const float* h0    = (const float*)d_in[0];
  const float* h1    = (const float*)d_in[1];
  const float* uph0  = (const float*)d_in[2];
  const float* uph1  = (const float*)d_in[3];
  const float* x     = (const float*)d_in[4];
  const float* xprev = (const float*)d_in[5];
  const float* r     = (const float*)d_in[6];
  const float* b00   = (const float*)d_in[7];
  const float* b01   = (const float*)d_in[8];
  const float* b10   = (const float*)d_in[9];
  const float* b11   = (const float*)d_in[10];
  const int*   esrc  = (const int*)  d_in[11];
  auto Pf = [&](int i) { return (const float*)d_in[i]; };

  float* ws    = (float*)d_ws;
  float* hs0   = ws;                                  // NT_*32
  float* hs1   = hs0 + (size_t)NT_*CCH;               // NT_*96
  float* msg_k = hs1 + (size_t)NT_*CCH*3;             // E*32
  float* msg_v = msg_k + (size_t)ETOT*32;             // E*32
  __bf16* wpool = (__bf16*)(msg_v + (size_t)ETOT*32); // packed weights (~213 KB)

  const int cnf[4] = {1, 1, 1, 3};
  EdgeArgs ea;
  PackArgs pa;
  size_t off = 0; int mi = 0; int total_tiles = 0;
  for (int vk = 0; vk < 2; ++vk)          // 12..51 = 'v', 52..91 = 'k'
    for (int c = 0; c < 4; ++c) {         // combos '00','01','10','11', 10 arrays each
      int base = 12 + vk*40 + c*10;
      RadialP& rp = ea.rp[vk][c];
      rp.w1 = Pf(base);     rp.b1  = Pf(base+1); rp.g1 = Pf(base+2); rp.be1 = Pf(base+3);
      rp.b2 = Pf(base+5);   rp.g2  = Pf(base+6); rp.be2 = Pf(base+7);
      rp.b3 = Pf(base+9);
      // w2: [32,32] -> 2 column tiles
      pa.d[mi] = PackDesc{ Pf(base+4), 32, 2, (unsigned long)off };
      rp.w2p = wpool + off; off += (size_t)2*32*16; total_tiles += 2; ++mi;
      // w3: [32, 256*nf] -> (256*nf)/16 column tiles
      int OUT = 256*cnf[c], nt3 = OUT/16;
      pa.d[mi] = PackDesc{ Pf(base+8), OUT, nt3, (unsigned long)off };
      rp.w3p = wpool + off; off += (size_t)nt3*32*16; total_tiles += nt3; ++mi;
    }
  const float* qw0 = Pf(92); const float* qw1 = Pf(93);
  const float* pw0 = Pf(94); const float* pw1 = Pf(95);
  const float* ng0 = Pf(96); const float* nb0 = Pf(97);
  const float* ng1 = Pf(98); const float* nb1 = Pf(99);

  pack_weights_kernel<<<total_tiles, 32, 0, stream>>>(pa, wpool);
  interp_kernel<<<NT_/128, 128, 0, stream>>>(h0, h1, uph0, uph1, x, xprev, hs0, hs1);
  edge_kernel<<<ETOT/TE, 32, 0, stream>>>(hs0, hs1, r, b00, b01, b10, b11, esrc,
                                          msg_v, msg_k, ea);
  node_kernel<<<NT_/4, 128, 0, stream>>>(hs0, hs1, msg_v, msg_k, qw0, qw1, pw0, pw1,
                                         ng0, nb0, ng1, nb1, (float*)d_out);
}
